// CausalSelfAttention_12678743458361
// MI455X (gfx1250) — matrix operand order repaired and checked by reading, not compiled, not run on hardware
//
#include <hip/hip_runtime.h>
#include <hip/hip_bf16.h>

// ---------------------------------------------------------------------------
// CDNA5 (gfx1250) causal self-attention: bf16 WMMA GEMMs + flash attention,
// async global->LDS tile staging (ASYNCcnt) with LDS double buffering.
// ---------------------------------------------------------------------------

typedef __attribute__((ext_vector_type(16))) __bf16 v16bf;
typedef __attribute__((ext_vector_type(8)))  float  v8f;

union AFrag { v16bf v; unsigned u[8]; };
union BFrag { v16bf v; unsigned u[8]; };
union Facc  { v8f   v; float    f[8]; };

__device__ __forceinline__ unsigned short f2bf(float f) {
  unsigned u = __builtin_bit_cast(unsigned, f);
  unsigned r = u + 0x7FFFu + ((u >> 16) & 1u);   // round-to-nearest-even
  return (unsigned short)(r >> 16);
}

__device__ __forceinline__ v8f wmma_bf16(v16bf a, v16bf b, v8f c) {
  // emits v_wmma_f32_16x16x32_bf16
  return __builtin_amdgcn_wmma_f32_16x16x32_bf16(false, a, false, b,
                                                 (short)0, c, false, false);
}

// LDS offset (low 32 bits of generic pointer == wave-relative LDS address)
__device__ __forceinline__ unsigned lds_off(const void* p) {
  return (unsigned)(uintptr_t)p;
}

// 16B async copy global -> LDS (tracked by ASYNCcnt)
__device__ __forceinline__ void async_b128(unsigned ldsoff, const void* gaddr) {
  asm volatile("global_load_async_to_lds_b128 %0, %1, off"
               :: "v"(ldsoff), "v"(gaddr) : "memory");
}

// ------------------------- elementwise converters --------------------------

__global__ void cvt_bf16_kernel(const float* __restrict__ in,
                                unsigned short* __restrict__ out, int n) {
  for (int i = blockIdx.x * blockDim.x + threadIdx.x; i < n;
       i += gridDim.x * blockDim.x)
    out[i] = f2bf(in[i]);
}

// W [1024][1024] f32 row-major -> Wt [1024][1024] bf16 (transposed: [N][K])
__global__ void transpose1024_bf16_kernel(const float* __restrict__ in,
                                          unsigned short* __restrict__ out) {
  const int total = 1024 * 1024;
  for (int i = blockIdx.x * blockDim.x + threadIdx.x; i < total;
       i += gridDim.x * blockDim.x) {
    int r = i >> 10;        // k
    int c = i & 1023;       // n
    out[((size_t)c << 10) + r] = f2bf(in[i]);
  }
}

// ------------------------------ WMMA GEMM ----------------------------------
// C[M,N] = A[M,K](bf16) * Bt[N,K](bf16)^T + bias
// outF != nullptr : write f32 row-major [M][N]
// else            : write bf16 split-head [B=4,H=16,T=2048,Dh=64]
__global__ __launch_bounds__(256) void gemm_bf16_kernel(
    const unsigned short* __restrict__ A,
    const unsigned short* __restrict__ Bt,
    const float* __restrict__ bias,
    unsigned short* __restrict__ outS,
    float* __restrict__ outF,
    int M, int N, int K) {
  __shared__ alignas(16) unsigned short As[2][128 * 40];  // [m][k] pad 40
  __shared__ alignas(16) unsigned short Bs[2][128 * 40];  // [n][k] pad 40

  const int tid  = threadIdx.x;
  const int wave = tid >> 5;
  const int lane = tid & 31;
  const int half = lane >> 4;
  const int l16  = lane & 15;
  const int wm   = wave >> 1;   // 0..3
  const int wn   = wave & 1;    // 0..1
  const int m0   = blockIdx.y * 128;
  const int n0   = blockIdx.x * 128;

  Facc acc[2][4];
  for (int mt = 0; mt < 2; ++mt)
    for (int nt = 0; nt < 4; ++nt)
      for (int j = 0; j < 8; ++j) acc[mt][nt].f[j] = 0.f;

  const int row  = tid >> 1;
  const int part = tid & 1;

  const unsigned short* Ab = A  + (size_t)(m0 + row) * K + part * 16;
  const unsigned short* Bb = Bt + (size_t)(n0 + row) * K + part * 16;
  unsigned aoff[2], boff[2];
  for (int p = 0; p < 2; ++p) {
    aoff[p] = lds_off(&As[p][row * 40 + part * 16]);
    boff[p] = lds_off(&Bs[p][row * 40 + part * 16]);
  }

  // prologue: stage k0 = 0 into buffer 0
  async_b128(aoff[0],      Ab);
  async_b128(aoff[0] + 16, Ab + 8);
  async_b128(boff[0],      Bb);
  async_b128(boff[0] + 16, Bb + 8);

  int p = 0;
  for (int k0 = 0; k0 < K; k0 += 32, p ^= 1) {
    const int kn = k0 + 32;
    if (kn < K) {        // stage next tile into the other buffer
      async_b128(aoff[1 - p],      Ab + kn);
      async_b128(aoff[1 - p] + 16, Ab + kn + 8);
      async_b128(boff[1 - p],      Bb + kn);
      async_b128(boff[1 - p] + 16, Bb + kn + 8);
      // async loads complete in order: cnt<=4 => current tile landed
      asm volatile("s_wait_asynccnt 0x4" ::: "memory");
    } else {
      asm volatile("s_wait_asynccnt 0x0" ::: "memory");
    }
    __syncthreads();

    const unsigned short* Asp = As[p];
    const unsigned short* Bsp = Bs[p];
    AFrag af[2];
    BFrag bf[4];
    for (int mt = 0; mt < 2; ++mt) {
      const int r = wm * 32 + mt * 16 + l16;
      for (int i = 0; i < 8; ++i) {
        const int kk = half * 16 + 2 * i;
        af[mt].u[i] = *(const unsigned*)&Asp[r * 40 + kk];
      }
    }
    for (int nt = 0; nt < 4; ++nt) {
      const int c = wn * 64 + nt * 16 + l16;
      for (int i = 0; i < 8; ++i)
        bf[nt].u[i] = *(const unsigned*)&Bsp[c * 40 + half * 16 + 2 * i];
    }
    for (int mt = 0; mt < 2; ++mt)
      for (int nt = 0; nt < 4; ++nt)
        acc[mt][nt].v = wmma_bf16(af[mt].v, bf[nt].v, acc[mt][nt].v);

    __syncthreads();   // all waves done reading buffer p before it is refilled
  }

  // epilogue
  for (int mt = 0; mt < 2; ++mt)
    for (int nt = 0; nt < 4; ++nt)
      for (int j = 0; j < 8; ++j) {
        const int r = m0 + wm * 32 + mt * 16 + j + 8 * half;
        const int c = n0 + wn * 64 + nt * 16 + l16;
        const float v = acc[mt][nt].f[j] + bias[c];
        if (outF) {
          outF[(size_t)r * N + c] = v;
        } else {
          const int b = r >> 11, t = r & 2047;   // T = 2048
          const int h = c >> 6,  d = c & 63;     // Dh = 64
          outS[(((size_t)(b * 16 + h)) * 2048 + t) * 64 + d] = f2bf(v);
        }
      }
}

// --------------------------- flash attention -------------------------------
// Q,K,V bf16 [B,H,T,64]; Y bf16 [B,T,1024]. 128 threads = 4 waves, each wave
// owns 16 query rows; key blocks of 32 staged in LDS; online softmax.
__global__ __launch_bounds__(128) void attn_kernel(
    const unsigned short* __restrict__ Qg,
    const unsigned short* __restrict__ Kg,
    const unsigned short* __restrict__ Vg,
    unsigned short* __restrict__ Yg) {
  constexpr int TSEQ = 2048;
  constexpr int DH   = 64;

  __shared__ alignas(16) unsigned short Ks[32 * 72];       // [key][dh]
  __shared__ alignas(16) unsigned short Vs[64 * 40];       // [dh][key]
  __shared__ alignas(16) unsigned short Ps[4][16 * 40];    // per-wave P tile

  const int tid  = threadIdx.x;
  const int wave = tid >> 5;
  const int lane = tid & 31;
  const int half = lane >> 4;
  const int l16  = lane & 15;

  const int bh = blockIdx.x >> 5;   // b*16 + h
  const int qb = blockIdx.x & 31;
  const int q0 = qb * 64;
  const int b  = bh >> 4;
  const int h  = bh & 15;
  const size_t base = (size_t)bh * TSEQ * DH;

  // loop-invariant Q fragments (two Dh chunks of 32)
  AFrag qf[2];
  {
    const unsigned short* qrow = Qg + base + (size_t)(q0 + wave * 16 + l16) * DH;
    for (int c = 0; c < 2; ++c)
      for (int i = 0; i < 8; ++i) {
        const int dh = c * 32 + half * 16 + 2 * i;
        qf[c].u[i] = *(const unsigned*)(qrow + dh);
      }
  }

  Facc acc[4];
  for (int t = 0; t < 4; ++t)
    for (int j = 0; j < 8; ++j) acc[t].f[j] = 0.f;
  float m[8], l[8];
  for (int j = 0; j < 8; ++j) { m[j] = -1e30f; l[j] = 0.f; }

  const float kScale = 0.125f * 1.44269504f;   // 1/sqrt(64) * log2(e)
  const int nkb = (q0 >> 5) + 2;               // key blocks covering k <= q0+63

  const int key = tid >> 2;   // cooperative tile-load role
  const int ch  = tid & 3;
  const unsigned koff = lds_off(&Ks[key * 72 + ch * 16]);

  for (int ib = 0; ib < nkb; ++ib) {
    const int kb = ib * 32;
    __syncthreads();
    {
      // K tile: async global->LDS (row-major [key][dh])
      const unsigned short* krow = Kg + base + (size_t)(kb + key) * DH + ch * 16;
      async_b128(koff,      krow);
      async_b128(koff + 16, krow + 8);
      // V tile: through VGPRs, transposed into LDS [dh][key]
      const unsigned short* vrow = Vg + base + (size_t)(kb + key) * DH + ch * 16;
      union { uint4 q; unsigned short s[8]; } p0, p1;
      p0.q = *(const uint4*)(vrow);
      p1.q = *(const uint4*)(vrow + 8);
      for (int j = 0; j < 8; ++j) Vs[(ch * 16 + j) * 40 + key]     = p0.s[j];
      for (int j = 0; j < 8; ++j) Vs[(ch * 16 + 8 + j) * 40 + key] = p1.s[j];
      if (ib + 1 < nkb) {
        __builtin_prefetch(krow + 32 * DH, 0, 1);
        __builtin_prefetch(vrow + 32 * DH, 0, 1);
      }
      asm volatile("s_wait_asynccnt 0x0" ::: "memory");
    }
    __syncthreads();

    // S = Q K^T  (two 16-key tiles, Dh split into 2 x K=32 WMMAs each)
    Facc s[2];
    for (int t = 0; t < 2; ++t) {
      BFrag kf0, kf1;
      const int ky = t * 16 + l16;
      for (int i = 0; i < 8; ++i) {
        kf0.u[i] = *(const unsigned*)&Ks[ky * 72 + 0  + half * 16 + 2 * i];
        kf1.u[i] = *(const unsigned*)&Ks[ky * 72 + 32 + half * 16 + 2 * i];
      }
      v8f z;
      for (int i = 0; i < 8; ++i) z[i] = 0.f;
      v8f sv = wmma_bf16(qf[0].v, kf0.v, z);
      sv     = wmma_bf16(qf[1].v, kf1.v, sv);
      s[t].v = sv;
    }

    // scale + causal mask + online softmax stats
    float alpha[8];
    for (int j = 0; j < 8; ++j) {
      const int qg = q0 + wave * 16 + j + 8 * half;
      const int k0g = kb + l16;
      s[0].f[j] = (k0g      <= qg) ? s[0].f[j] * kScale : -1e30f;
      s[1].f[j] = (k0g + 16 <= qg) ? s[1].f[j] * kScale : -1e30f;

      float r = fmaxf(s[0].f[j], s[1].f[j]);
      for (int o = 1; o < 16; o <<= 1) r = fmaxf(r, __shfl_xor(r, o, 16));
      const float mn = fmaxf(m[j], r);
      const float al = exp2f(m[j] - mn);
      const float p0 = exp2f(s[0].f[j] - mn);
      const float p1 = exp2f(s[1].f[j] - mn);
      s[0].f[j] = p0;
      s[1].f[j] = p1;
      float rs = p0 + p1;
      for (int o = 1; o < 16; o <<= 1) rs += __shfl_xor(rs, o, 16);
      l[j] = l[j] * al + rs;
      m[j] = mn;
      alpha[j] = al;
    }
    for (int t = 0; t < 4; ++t)
      for (int j = 0; j < 8; ++j) acc[t].f[j] *= alpha[j];

    // P (C-layout) -> LDS -> A-fragment layout
    for (int j = 0; j < 8; ++j) {
      const int r = j + 8 * half;
      Ps[wave][r * 40 + l16]      = f2bf(s[0].f[j]);
      Ps[wave][r * 40 + 16 + l16] = f2bf(s[1].f[j]);
    }
    __syncthreads();
    AFrag pf;
    for (int i = 0; i < 8; ++i) {
      const int kk = half * 16 + 2 * i;
      pf.u[i] = *(const unsigned*)&Ps[wave][l16 * 40 + kk];
    }

    // O += P @ V
    for (int dt = 0; dt < 4; ++dt) {
      BFrag vf;
      const int dh = dt * 16 + l16;
      for (int i = 0; i < 8; ++i)
        vf.u[i] = *(const unsigned*)&Vs[dh * 40 + half * 16 + 2 * i];
      acc[dt].v = wmma_bf16(pf.v, vf.v, acc[dt].v);
    }
  }

  // epilogue: divide by l, write bf16 [B,T,1024]
  for (int dt = 0; dt < 4; ++dt)
    for (int j = 0; j < 8; ++j) {
      const int tq  = q0 + wave * 16 + j + 8 * half;
      const int col = h * 64 + dt * 16 + l16;
      Yg[((size_t)b * TSEQ + tq) * 1024 + col] = f2bf(acc[dt].f[j] / l[j]);
    }
}

// ------------------------------- host side ---------------------------------

extern "C" void kernel_launch(void* const* d_in, const int* in_sizes, int n_in,
                              void* d_out, int out_size, void* d_ws, size_t ws_size,
                              hipStream_t stream) {
  const float* x  = (const float*)d_in[0];
  const float* Wq = (const float*)d_in[1];
  const float* bq = (const float*)d_in[2];
  const float* Wk = (const float*)d_in[3];
  const float* bk = (const float*)d_in[4];
  const float* Wv = (const float*)d_in[5];
  const float* bv = (const float*)d_in[6];
  const float* Wp = (const float*)d_in[7];
  const float* bp = (const float*)d_in[8];
  float* out = (float*)d_out;

  char* ws = (char*)d_ws;
  unsigned short* xb   = (unsigned short*)(ws);                 // 16 MB
  unsigned short* wqt  = (unsigned short*)(ws + (16u << 20));   // 2 MB each
  unsigned short* wkt  = (unsigned short*)(ws + (18u << 20));
  unsigned short* wvt  = (unsigned short*)(ws + (20u << 20));
  unsigned short* wpt  = (unsigned short*)(ws + (22u << 20));
  unsigned short* Qb   = (unsigned short*)(ws + (24u << 20));   // 16 MB each
  unsigned short* Kb   = (unsigned short*)(ws + (40u << 20));
  unsigned short* Vb   = (unsigned short*)(ws + (56u << 20));
  unsigned short* yatt = (unsigned short*)(ws + (72u << 20));   // 16 MB

  const int M = 8192, N = 1024, K = 1024;

  cvt_bf16_kernel<<<2048, 256, 0, stream>>>(x, xb, M * K);
  transpose1024_bf16_kernel<<<2048, 256, 0, stream>>>(Wq, wqt);
  transpose1024_bf16_kernel<<<2048, 256, 0, stream>>>(Wk, wkt);
  transpose1024_bf16_kernel<<<2048, 256, 0, stream>>>(Wv, wvt);
  transpose1024_bf16_kernel<<<2048, 256, 0, stream>>>(Wp, wpt);

  dim3 gg(N / 128, M / 128), gb(256);
  gemm_bf16_kernel<<<gg, gb, 0, stream>>>(xb, wqt, bq, Qb, nullptr, M, N, K);
  gemm_bf16_kernel<<<gg, gb, 0, stream>>>(xb, wkt, bk, Kb, nullptr, M, N, K);
  gemm_bf16_kernel<<<gg, gb, 0, stream>>>(xb, wvt, bv, Vb, nullptr, M, N, K);

  attn_kernel<<<2048, 128, 0, stream>>>(Qb, Kb, Vb, yatt);

  gemm_bf16_kernel<<<gg, gb, 0, stream>>>(yatt, wpt, bp, nullptr, out, M, N, K);
}
